// AblationCausalSelfAttention_39470749450268
// MI455X (gfx1250) — matrix operand order repaired and checked by reading, not compiled, not run on hardware
//
#include <hip/hip_runtime.h>

// ===========================================================================
// AblationCausalSelfAttention for MI455X (gfx1250, wave32, WMMA bf16 + TDM).
//
//   1) qkv_gemm_kernel : qkv = x @ W_attn + b_attn   (fp32 -> bf16 on the fly)
//                        writes q,k as [B,H,T,D] and v transposed [B,H,D,T]
//   2) attn_kernel     : per (b,h, 16-row block): 3 sweeps over columns
//                        a) row mean of raw scores (unmasked, per reference)
//                        b) mean-abs-deviation of centered scores
//                        c) causal sweep: w = relu((s-mean)*gain/mad)^4,
//                           y += w @ v, rowsum += w;  y /= rowsum + 1e-6
//                        writes y as bf16 [B*T, C]
//   3) proj_gemm_kernel: out = y @ W_proj + b_proj  (fp32 output); the bf16
//                        A tile is staged into LDS by the Tensor Data Mover.
// ===========================================================================

#if defined(__HIP_DEVICE_COMPILE__) && defined(__has_builtin)
#if __has_builtin(__builtin_amdgcn_tensor_load_to_lds) && \
    __has_builtin(__builtin_amdgcn_s_wait_tensorcnt)
#define HAVE_TDM 1
#endif
#endif
#ifndef HAVE_TDM
#define HAVE_TDM 0
#endif

namespace {

constexpr int T  = 2048;
constexpr int C  = 1024;
constexpr int H  = 16;
constexpr int D  = 64;
constexpr int BT = 2 * T;   // B*T = 4096
constexpr int N3 = 3 * C;   // 3072

typedef __attribute__((ext_vector_type(16))) __bf16 v16bf;
typedef __attribute__((ext_vector_type(8)))  float  v8f;
typedef __attribute__((ext_vector_type(4)))  unsigned int uint4v;
typedef __attribute__((ext_vector_type(8)))  int    int8v;
typedef __attribute__((ext_vector_type(4)))  int    int4v;

union Frag {
  v16bf  v;
  __bf16 h[16];
  int4   q2[2];
};

__device__ __forceinline__ v8f wmma_bf16(v16bf a, v16bf b, v8f c) {
  // (neg_a, A, neg_b, B, c_mod, C, reuse_a, reuse_b)
  return __builtin_amdgcn_wmma_f32_16x16x32_bf16(false, a, false, b,
                                                 (short)0, c, false, false);
}

__device__ __forceinline__ v8f vzero() {
  v8f z = {0.f, 0.f, 0.f, 0.f, 0.f, 0.f, 0.f, 0.f};
  return z;
}

// Issue one TDM 2D-tile load (global -> LDS). Descriptor packing per
// CDNA5 ISA (D# group0/group1). All args uniform.
__device__ __forceinline__ void tdm_load_tile_2d(unsigned int lds_off,
                                                 const void* gptr,
                                                 unsigned int tensor_d0,
                                                 unsigned int tensor_d1,
                                                 unsigned int tile_d0,
                                                 unsigned int tile_d1,
                                                 unsigned int stride_d0) {
#if HAVE_TDM
  const unsigned long long ga = (unsigned long long)(uintptr_t)gptr;
  uint4v g0;
  g0.x = 1u;                                     // count=1, user descriptor
  g0.y = lds_off;                                // lds_addr [63:32]
  g0.z = (unsigned int)(ga & 0xFFFFFFFFu);       // global_addr [95:64]
  g0.w = (unsigned int)((ga >> 32) & 0x1FFFFFFu) // global_addr [120:96]
         | (2u << 30);                           // type=2 ("image")
  int8v g1;
  g1[0] = 0x00010000;                            // data_size=1 -> 2 bytes
  g1[1] = (int)((tensor_d0 & 0xFFFFu) << 16);    // tensor_dim0[15:0]
  g1[2] = (int)(((tensor_d0 >> 16) & 0xFFFFu) |
                ((tensor_d1 & 0xFFFFu) << 16));  // dim0[31:16] | dim1[15:0]
  g1[3] = (int)(((tensor_d1 >> 16) & 0xFFFFu) |
                (tile_d0 << 16));                // dim1[31:16] | tile_dim0
  g1[4] = (int)(tile_d1 & 0xFFFFu);              // tile_dim1 | tile_dim2=0
  g1[5] = (int)stride_d0;                        // tensor_dim0_stride[31:0]
  g1[6] = 0;                                     // stride hi | dim1_stride lo
  g1[7] = 0;
  int4v z4 = {0, 0, 0, 0};
#if __clang_major__ >= 23
  int8v z8 = {0, 0, 0, 0, 0, 0, 0, 0};
  __builtin_amdgcn_tensor_load_to_lds(g0, g1, z4, z4, z8, 0);
#else
  __builtin_amdgcn_tensor_load_to_lds(g0, g1, z4, z4, 0);
#endif
#else
  (void)lds_off; (void)gptr; (void)tensor_d0; (void)tensor_d1;
  (void)tile_d0; (void)tile_d1; (void)stride_d0;
#endif
}

// ---------------------------------------------------------------------------
// Kernel 1: qkv = x @ W_attn + b_attn ; M=4096, N=3072, K=1024
// 256 threads (8 waves), block tile 64x128, wave tile 32x32, BK=32.
// ---------------------------------------------------------------------------
__global__ __launch_bounds__(256)
void qkv_gemm_kernel(const float* __restrict__ x, const float* __restrict__ Wa,
                     const float* __restrict__ ba,
                     __bf16* __restrict__ qbf, __bf16* __restrict__ kbf,
                     __bf16* __restrict__ vtr)
{
  __shared__ __bf16 As[64][32];    // [m][k]
  __shared__ __bf16 Bs[128][32];   // transposed: [n][k]

  const int m0    = blockIdx.x * 64;
  const int n0    = blockIdx.y * 128;
  const int tid   = threadIdx.x;
  const int lane  = tid & 31;
  const int wave  = tid >> 5;
  const int nlane = lane & 15;
  const int half  = lane >> 4;
  const int wm    = (wave >> 2) * 32;
  const int wn    = (wave & 3) * 32;

  v8f acc[2][2];
  acc[0][0] = vzero(); acc[0][1] = vzero();
  acc[1][0] = vzero(); acc[1][1] = vzero();

  const int arow = (tid * 8) >> 5;
  const int acol = (tid * 8) & 31;
  const int bkk  = tid >> 3;
  const int bnb  = (tid & 7) * 16;

  for (int k0 = 0; k0 < C; k0 += 32) {
    // Stage A tile: 64x32 fp32 -> bf16, 8 contiguous elements / thread.
    {
      const float* src = x + (size_t)(m0 + arow) * C + k0 + acol;
      float4 f0 = ((const float4*)src)[0];
      float4 f1 = ((const float4*)src)[1];
      __bf16* dst = &As[arow][acol];
      dst[0] = (__bf16)f0.x; dst[1] = (__bf16)f0.y;
      dst[2] = (__bf16)f0.z; dst[3] = (__bf16)f0.w;
      dst[4] = (__bf16)f1.x; dst[5] = (__bf16)f1.y;
      dst[6] = (__bf16)f1.z; dst[7] = (__bf16)f1.w;
    }
    // Stage B tile transposed: W[k][n] -> Bs[n][k], 16 elements / thread.
    {
      const float* src = Wa + (size_t)(k0 + bkk) * N3 + n0 + bnb;
      #pragma unroll
      for (int i = 0; i < 16; i += 4) {
        float4 f = ((const float4*)(src + i))[0];
        Bs[bnb + i + 0][bkk] = (__bf16)f.x;
        Bs[bnb + i + 1][bkk] = (__bf16)f.y;
        Bs[bnb + i + 2][bkk] = (__bf16)f.z;
        Bs[bnb + i + 3][bkk] = (__bf16)f.w;
      }
    }
    if (k0 + 32 < C) {  // hint next K tiles into cache
      __builtin_prefetch(x + (size_t)(m0 + arow) * C + k0 + 32 + acol, 0, 1);
      __builtin_prefetch(Wa + (size_t)(k0 + 32 + bkk) * N3 + n0 + bnb, 0, 1);
    }
    __syncthreads();

    Frag af[2], bfr[2];
    #pragma unroll
    for (int t2 = 0; t2 < 2; ++t2) {
      // A fragment (16x32): lane m = nlane; k groups at 8*half and 16+8*half.
      const __bf16* ap = &As[wm + t2 * 16 + nlane][0];
      af[t2].q2[0] = *(const int4*)(ap + 8 * half);
      af[t2].q2[1] = *(const int4*)(ap + 16 + 8 * half);
      // B fragment (32x16): lane n = nlane; k groups at 8*half and 16+8*half.
      const __bf16* bp = &Bs[wn + t2 * 16 + nlane][0];
      bfr[t2].q2[0] = *(const int4*)(bp + 8 * half);
      bfr[t2].q2[1] = *(const int4*)(bp + 16 + 8 * half);
    }
    #pragma unroll
    for (int tm = 0; tm < 2; ++tm)
      #pragma unroll
      for (int tn = 0; tn < 2; ++tn)
        acc[tm][tn] = wmma_bf16(af[tm].v, bfr[tn].v, acc[tm][tn]);
    __syncthreads();
  }

  // Epilogue: +bias, split q/k/v (seg is UNIFORM per block: 128 | 1024),
  // head reshape; v stored transposed [B,H,D,T].
  const int seg = n0 >> 10;   // 0:q 1:k 2:v — uniform branch
  #pragma unroll
  for (int tm = 0; tm < 2; ++tm) {
    #pragma unroll
    for (int tn = 0; tn < 2; ++tn) {
      const int n    = n0 + wn + tn * 16 + nlane;
      const float bs = ba[n];
      const int cn   = n & (C - 1);
      const int hh   = cn >> 6;
      const int dd   = cn & (D - 1);
      if (seg == 0) {
        #pragma unroll
        for (int r = 0; r < 8; ++r) {
          const int m  = m0 + wm + tm * 16 + r + 8 * half;
          const int bb = m >> 11;
          const int tt = m & (T - 1);
          qbf[(((size_t)bb * H + hh) * T + tt) * D + dd] =
              (__bf16)(acc[tm][tn][r] + bs);
        }
      } else if (seg == 1) {
        #pragma unroll
        for (int r = 0; r < 8; ++r) {
          const int m  = m0 + wm + tm * 16 + r + 8 * half;
          const int bb = m >> 11;
          const int tt = m & (T - 1);
          kbf[(((size_t)bb * H + hh) * T + tt) * D + dd] =
              (__bf16)(acc[tm][tn][r] + bs);
        }
      } else {
        #pragma unroll
        for (int r = 0; r < 8; ++r) {
          const int m  = m0 + wm + tm * 16 + r + 8 * half;
          const int bb = m >> 11;
          const int tt = m & (T - 1);
          vtr[(((size_t)bb * H + hh) * D + dd) * T + tt] =
              (__bf16)(acc[tm][tn][r] + bs);
        }
      }
    }
  }
}

// ---------------------------------------------------------------------------
// Kernel 2: attention for one (b,h) and one 16-row block, 1 wave / workgroup.
// ---------------------------------------------------------------------------
__global__ __launch_bounds__(32)
void attn_kernel(const __bf16* __restrict__ qbf, const __bf16* __restrict__ kbf,
                 const __bf16* __restrict__ vtr, const float* __restrict__ gainp,
                 __bf16* __restrict__ ybf)
{
  __shared__ __bf16 Ws[16][32];   // score-weight bounce: C-layout -> A-layout

  const int lane  = threadIdx.x;
  const int nlane = lane & 15;
  const int half  = lane >> 4;
  const int m0    = blockIdx.x * 16;
  const int bh    = blockIdx.y;          // b*H + h

  const __bf16* qp = qbf + ((size_t)bh * T + m0) * D;
  const __bf16* kp = kbf + (size_t)bh * T * D;
  const __bf16* vp = vtr + (size_t)bh * D * T;
  const float gain = *gainp;

  // q fragments: 16x64 = two 16x32 A-fragments, kept in registers all kernel.
  Frag qa[2];
  #pragma unroll
  for (int f = 0; f < 2; ++f) {
    const __bf16* p = qp + (size_t)nlane * D + f * 32;
    qa[f].q2[0] = *(const int4*)(p + 16 * half);
    qa[f].q2[1] = *(const int4*)(p + 16 * half + 8);
  }

  // Compute two 16x16 score tiles for columns [j0, j0+32): s = q . k^T.
  auto score32 = [&](int j0, v8f& s0, v8f& s1) {
    s0 = vzero(); s1 = vzero();
    {
      const __bf16* p = kp + (size_t)(j0 + nlane) * D;   // B frag: n = nlane
      Frag k0f, k1f;
      k0f.q2[0] = *(const int4*)(p + 16 * half);
      k0f.q2[1] = *(const int4*)(p + 16 * half + 8);
      k1f.q2[0] = *(const int4*)(p + 32 + 16 * half);
      k1f.q2[1] = *(const int4*)(p + 32 + 16 * half + 8);
      s0 = wmma_bf16(qa[0].v, k0f.v, s0);
      s0 = wmma_bf16(qa[1].v, k1f.v, s0);
    }
    {
      const __bf16* p = kp + (size_t)(j0 + 16 + nlane) * D;
      Frag k0f, k1f;
      k0f.q2[0] = *(const int4*)(p + 16 * half);
      k0f.q2[1] = *(const int4*)(p + 16 * half + 8);
      k1f.q2[0] = *(const int4*)(p + 32 + 16 * half);
      k1f.q2[1] = *(const int4*)(p + 32 + 16 * half + 8);
      s1 = wmma_bf16(qa[0].v, k0f.v, s1);
      s1 = wmma_bf16(qa[1].v, k1f.v, s1);
    }
  };

  // ---- Pass 1a: row sums of raw (unmasked) scores -> mean ----
  float ssum[8];
  #pragma unroll
  for (int r = 0; r < 8; ++r) ssum[r] = 0.f;
  for (int j0 = 0; j0 < T; j0 += 32) {
    v8f s0, s1;
    score32(j0, s0, s1);
    #pragma unroll
    for (int r = 0; r < 8; ++r) ssum[r] += s0[r] + s1[r];
  }
  #pragma unroll
  for (int r = 0; r < 8; ++r)
    #pragma unroll
    for (int off = 1; off < 16; off <<= 1)
      ssum[r] += __shfl_xor(ssum[r], off, 32);
  float mean[8];
  #pragma unroll
  for (int r = 0; r < 8; ++r) mean[r] = ssum[r] * (1.0f / (float)T);

  // ---- Pass 1b: mean abs deviation of centered scores -> scale ----
  float sabs[8];
  #pragma unroll
  for (int r = 0; r < 8; ++r) sabs[r] = 0.f;
  for (int j0 = 0; j0 < T; j0 += 32) {
    v8f s0, s1;
    score32(j0, s0, s1);
    #pragma unroll
    for (int r = 0; r < 8; ++r)
      sabs[r] += fabsf(s0[r] - mean[r]) + fabsf(s1[r] - mean[r]);
  }
  #pragma unroll
  for (int r = 0; r < 8; ++r)
    #pragma unroll
    for (int off = 1; off < 16; off <<= 1)
      sabs[r] += __shfl_xor(sabs[r], off, 32);
  float scl[8];
  #pragma unroll
  for (int r = 0; r < 8; ++r)
    scl[r] = gain / (sabs[r] * (1.0f / (float)T) + 1e-6f);

  // ---- Pass 2: causal sweep: w = relu(s_norm)^4, y += w @ v ----
  v8f yacc[4];
  #pragma unroll
  for (int nt = 0; nt < 4; ++nt) yacc[nt] = vzero();
  float rsum[8];
  #pragma unroll
  for (int r = 0; r < 8; ++r) rsum[r] = 0.f;

  for (int j0 = 0; j0 < m0 + 16; j0 += 32) {
    v8f s0, s1;
    score32(j0, s0, s1);
    #pragma unroll
    for (int jt = 0; jt < 2; ++jt) {
      #pragma unroll
      for (int r = 0; r < 8; ++r) {
        const int j = j0 + jt * 16 + nlane;
        const int i = m0 + r + 8 * half;
        float sv = (jt == 0) ? s0[r] : s1[r];
        float tv = (sv - mean[r]) * scl[r];
        tv = (j <= i) ? fmaxf(tv, 0.f) : 0.f;
        const float t2 = tv * tv;
        const float w  = t2 * t2;                    // relu^4
        rsum[r] += w;
        Ws[r + 8 * half][jt * 16 + nlane] = (__bf16)w;  // C-layout -> [m][j]
      }
    }
    __syncthreads();   // single-wave WG: ordering fence for the LDS bounce

    // Re-load the 16x32 weight tile in A-fragment layout.
    Frag wa;
    const __bf16* wp = &Ws[nlane][0];
    wa.q2[0] = *(const int4*)(wp + 16 * half);
    wa.q2[1] = *(const int4*)(wp + 16 * half + 8);

    // y[16 x 64] += w[16 x 32] @ v[32 x 64]; v pre-transposed -> contiguous.
    #pragma unroll
    for (int nt = 0; nt < 4; ++nt) {
      Frag vf;
      const __bf16* p = vp + (size_t)(nt * 16 + nlane) * T + j0 + 16 * half;
      vf.q2[0] = *(const int4*)p;
      vf.q2[1] = *(const int4*)(p + 8);
      yacc[nt] = wmma_bf16(wa.v, vf.v, yacc[nt]);
    }
    __syncthreads();
  }

  #pragma unroll
  for (int r = 0; r < 8; ++r)
    #pragma unroll
    for (int off = 1; off < 16; off <<= 1)
      rsum[r] += __shfl_xor(rsum[r], off, 32);

  // Store y as bf16 [B*T, C] for the projection GEMM.
  const int batch = bh >> 4;
  const int hh    = bh & (H - 1);
  #pragma unroll
  for (int nt = 0; nt < 4; ++nt) {
    #pragma unroll
    for (int r = 0; r < 8; ++r) {
      const int m = m0 + r + 8 * half;
      const float val = yacc[nt][r] / (rsum[r] + 1e-6f);
      ybf[((size_t)(batch * T + m)) * C + hh * D + nt * 16 + nlane] = (__bf16)val;
    }
  }
}

// ---------------------------------------------------------------------------
// Kernel 3: out = y @ W_proj + b_proj ; M=4096, N=1024, K=1024 (fp32 out).
// The bf16 A tile (64x32 from y[4096][1024]) is DMA'd into LDS by the TDM.
// ---------------------------------------------------------------------------
__global__ __launch_bounds__(256)
void proj_gemm_kernel(const __bf16* __restrict__ ybf, const float* __restrict__ Wp,
                      const float* __restrict__ bp, float* __restrict__ out)
{
  __shared__ __bf16 As[64][32];
  __shared__ __bf16 Bs[128][32];

  const int m0    = blockIdx.x * 64;
  const int n0    = blockIdx.y * 128;
  const int tid   = threadIdx.x;
  const int lane  = tid & 31;
  const int wave  = tid >> 5;
  const int nlane = lane & 15;
  const int half  = lane >> 4;
  const int wm    = (wave >> 2) * 32;
  const int wn    = (wave & 3) * 32;

  v8f acc[2][2];
  acc[0][0] = vzero(); acc[0][1] = vzero();
  acc[1][0] = vzero(); acc[1][1] = vzero();

  const int bkk = tid >> 3;
  const int bnb = (tid & 7) * 16;
#if HAVE_TDM
  const unsigned int lds_as = (unsigned int)(uintptr_t)&As[0][0];
#endif

  for (int k0 = 0; k0 < C; k0 += 32) {
#if HAVE_TDM
    // Async tensor DMA: 32x64-element bf16 tile of y -> LDS (wave 0 issues;
    // TDM ignores EXEC, so issue must be wave-unique).
    if (wave == 0) {
      tdm_load_tile_2d(lds_as, ybf + (size_t)m0 * C + k0,
                       /*tensor_d0=*/C, /*tensor_d1=*/BT,
                       /*tile_d0=*/32, /*tile_d1=*/64,
                       /*stride_d0=*/C);
    }
#else
    {
      const int idx = tid * 8;
      const int row = idx >> 5;
      const int col = idx & 31;
      *(int4*)&As[row][col] =
          *(const int4*)(ybf + (size_t)(m0 + row) * C + k0 + col);
    }
#endif
    {
      const float* src = Wp + (size_t)(k0 + bkk) * C + n0 + bnb;
      #pragma unroll
      for (int i = 0; i < 16; i += 4) {
        float4 f = ((const float4*)(src + i))[0];
        Bs[bnb + i + 0][bkk] = (__bf16)f.x;
        Bs[bnb + i + 1][bkk] = (__bf16)f.y;
        Bs[bnb + i + 2][bkk] = (__bf16)f.z;
        Bs[bnb + i + 3][bkk] = (__bf16)f.w;
      }
    }
    if (k0 + 32 < C)
      __builtin_prefetch(Wp + (size_t)(k0 + 32 + bkk) * C + n0 + bnb, 0, 1);
#if HAVE_TDM
    __builtin_amdgcn_s_wait_tensorcnt(0);  // TENSORcnt is per-wave; 0 elsewhere
#endif
    __syncthreads();

    Frag af[2], bfr[2];
    #pragma unroll
    for (int t2 = 0; t2 < 2; ++t2) {
      const __bf16* ap = &As[wm + t2 * 16 + nlane][0];
      af[t2].q2[0] = *(const int4*)(ap + 8 * half);
      af[t2].q2[1] = *(const int4*)(ap + 16 + 8 * half);
      const __bf16* bpp = &Bs[wn + t2 * 16 + nlane][0];
      bfr[t2].q2[0] = *(const int4*)(bpp + 8 * half);
      bfr[t2].q2[1] = *(const int4*)(bpp + 16 + 8 * half);
    }
    #pragma unroll
    for (int tm = 0; tm < 2; ++tm)
      #pragma unroll
      for (int tn = 0; tn < 2; ++tn)
        acc[tm][tn] = wmma_bf16(af[tm].v, bfr[tn].v, acc[tm][tn]);
    __syncthreads();
  }

  #pragma unroll
  for (int tm = 0; tm < 2; ++tm) {
    #pragma unroll
    for (int tn = 0; tn < 2; ++tn) {
      const int n    = n0 + wn + tn * 16 + nlane;
      const float bs = bp[n];
      #pragma unroll
      for (int r = 0; r < 8; ++r) {
        const int m = m0 + wm + tm * 16 + r + 8 * half;
        out[(size_t)m * C + n] = acc[tm][tn][r] + bs;
      }
    }
  }
}

} // namespace

// ---------------------------------------------------------------------------
extern "C" void kernel_launch(void* const* d_in, const int* in_sizes, int n_in,
                              void* d_out, int out_size, void* d_ws, size_t ws_size,
                              hipStream_t stream)
{
  (void)in_sizes; (void)n_in; (void)out_size; (void)ws_size;

  const float* x      = (const float*)d_in[0];
  const float* W_attn = (const float*)d_in[1];
  const float* b_attn = (const float*)d_in[2];
  const float* W_proj = (const float*)d_in[3];
  const float* b_proj = (const float*)d_in[4];
  const float* gain   = (const float*)d_in[5];

  // Workspace: q, k (both [B,H,T,D]), v transposed [B,H,D,T], y [B*T,C]
  // — each 8 MiB of bf16; 32 MiB total.
  char* ws = (char*)d_ws;
  const size_t seg = (size_t)2 * H * T * D * sizeof(__bf16);
  __bf16* qbf = (__bf16*)(ws + 0 * seg);
  __bf16* kbf = (__bf16*)(ws + 1 * seg);
  __bf16* vtr = (__bf16*)(ws + 2 * seg);
  __bf16* ybf = (__bf16*)(ws + 3 * seg);

  qkv_gemm_kernel<<<dim3(BT / 64, N3 / 128), 256, 0, stream>>>(
      x, W_attn, b_attn, qbf, kbf, vtr);
  attn_kernel<<<dim3(T / 16, 2 * H), 32, 0, stream>>>(
      qbf, kbf, vtr, gain, ybf);
  proj_gemm_kernel<<<dim3(BT / 64, C / 128), 256, 0, stream>>>(
      ybf, W_proj, b_proj, (float*)d_out);
}
